// RegimeDetector_63393717289138
// MI455X (gfx1250) — matrix operand order repaired and checked by reading, not compiled, not run on hardware
//
#include <hip/hip_runtime.h>
#include <math.h>

// ---------------- problem constants ----------------
#define BB   32
#define LL   2048
#define DD   512
#define LOUT 2049     // (2048 + 2*10 - 20) + 1
#define KW   20
#define HD   256      // D/2
#define RR   4

// ---------------- device types ----------------
typedef __attribute__((ext_vector_type(16))) __bf16   v16bf;
typedef __attribute__((ext_vector_type(8)))  float    v8f;
typedef __attribute__((ext_vector_type(8)))  unsigned v8u;
typedef __attribute__((ext_vector_type(4)))  unsigned v4u;

union FragU { v8u u8; v4u u4[2]; v16bf bf; };

__device__ __forceinline__ unsigned short bf_bits(float f) {
  unsigned u = __builtin_bit_cast(unsigned, f);
  u += 0x7FFFu + ((u >> 16) & 1u);          // round-to-nearest-even
  return (unsigned short)(u >> 16);
}
__device__ __forceinline__ v8f vz8() {
  v8f z = {0.f,0.f,0.f,0.f,0.f,0.f,0.f,0.f}; return z;
}
__device__ __forceinline__ v8u uz8() {
  v8u z = {0u,0u,0u,0u,0u,0u,0u,0u}; return z;
}
__device__ __forceinline__ void ld_frag(FragU& f, const unsigned short* p) {
  f.u4[0] = *(const v4u*)p;
  f.u4[1] = *(const v4u*)(p + 16);
}

#define WMMA_BF16(A,Bm,C) \
  __builtin_amdgcn_wmma_f32_16x16x32_bf16(false,(A),false,(Bm),(short)0,(C),false,false)

// ---------------- conversion kernels ----------------
__global__ void __launch_bounds__(256) cvt_f32_bf16_kernel(
    const float* __restrict__ in, unsigned short* __restrict__ out, int n4) {
  int i = blockIdx.x * blockDim.x + threadIdx.x;
  if (i >= n4) return;
  float4 f = *(const float4*)(in + (size_t)i * 4);
  uint2 pk;
  pk.x = (unsigned)bf_bits(f.x) | ((unsigned)bf_bits(f.y) << 16);
  pk.y = (unsigned)bf_bits(f.z) | ((unsigned)bf_bits(f.w) << 16);
  *(uint2*)(out + (size_t)i * 4) = pk;
}

// conv1_w (co, ci, k) f32 -> (co, k, ci) bf16  (ci contiguous for B-fragments)
__global__ void __launch_bounds__(256) cvt_w1_kernel(
    const float* __restrict__ w, unsigned short* __restrict__ wt) {
  int idx = blockIdx.x * blockDim.x + threadIdx.x;
  if (idx >= DD * KW * DD) return;
  int ci = idx & (DD - 1);
  int k  = (idx >> 9) % KW;
  int co = idx / (KW * DD);
  wt[idx] = bf_bits(w[((size_t)co * DD + ci) * KW + k]);
}

// ---------------- conv1 implicit-GEMM (WMMA bf16) ----------------
// Each wave: M=32 output positions x N=64 output channels.
// CHECK=false: interior tile (no halo / no store predicates) -- hot path.
template <bool CHECK>
__device__ __forceinline__ void conv1_tile(
    const unsigned short* __restrict__ xbatch,
    const unsigned short* __restrict__ w1t,
    const float* __restrict__ bias,
    unsigned short* __restrict__ hrow,   // hb + b*LOUT*DD
    int t0, int co0, int m, int half, int klo)
{
  v8f acc0[4], acc1[4];
  #pragma unroll
  for (int s = 0; s < 4; ++s) { acc0[s] = vz8(); acc1[s] = vz8(); }

  for (int k = 0; k < KW; ++k) {
    const int ta0 = t0 + m + k - 10;
    const bool va0 = !CHECK || ((unsigned)ta0        < (unsigned)LL);
    const bool va1 = !CHECK || ((unsigned)(ta0 + 16) < (unsigned)LL);
    // lane bases; A1 tile and B sub-tiles reached via constant displacements
    const unsigned short* arow = xbatch + (size_t)ta0 * DD + (half << 3);
    const unsigned short* wrow =
        w1t + (size_t)k * DD + (size_t)(co0 + m) * (KW * DD) + klo;
    #pragma unroll 2
    for (int ci0 = 0; ci0 < DD; ci0 += 32) {
      // ---- issue all 12 loads (distinct regs -> one clause, one wait) ----
      FragU a0, a1, f0, f1, f2, f3;
      const unsigned short* pA = arow + ci0;
      if (va0) { a0.u4[0] = *(const v4u*)(pA);
                 a0.u4[1] = *(const v4u*)(pA + 16); }
      else     { a0.u8 = uz8(); }
      if (va1) { a1.u4[0] = *(const v4u*)(pA + 16 * DD);
                 a1.u4[1] = *(const v4u*)(pA + 16 * DD + 16); }
      else     { a1.u8 = uz8(); }
      const unsigned short* pW = wrow + ci0;
      ld_frag(f0, pW);
      ld_frag(f1, pW + 16 * (KW * DD));
      ld_frag(f2, pW + 32 * (KW * DD));
      ld_frag(f3, pW + 48 * (KW * DD));
      // ---- 8 back-to-back WMMAs; each B fragment used by a0 then a1 so all four
      //      B fragments stay live -> loads cannot be re-serialized ----
      acc0[0] = WMMA_BF16(a0.bf, f0.bf, acc0[0]);
      acc0[1] = WMMA_BF16(a0.bf, f1.bf, acc0[1]);
      acc0[2] = WMMA_BF16(a0.bf, f2.bf, acc0[2]);
      acc0[3] = WMMA_BF16(a0.bf, f3.bf, acc0[3]);
      acc1[0] = WMMA_BF16(a1.bf, f0.bf, acc1[0]);
      acc1[1] = WMMA_BF16(a1.bf, f1.bf, acc1[1]);
      acc1[2] = WMMA_BF16(a1.bf, f2.bf, acc1[2]);
      acc1[3] = WMMA_BF16(a1.bf, f3.bf, acc1[3]);
    }
  }

  #pragma unroll
  for (int s = 0; s < 4; ++s) {
    const int   co = co0 + (s << 4) + m;
    const float bv = bias[co];
    #pragma unroll
    for (int j = 0; j < 8; ++j) {
      const int t = t0 + j + (half << 3);
      if (!CHECK || t < LOUT)
        hrow[(size_t)t * DD + co] = bf_bits(fmaxf(acc0[s][j] + bv, 0.f));
      const int t2 = t + 16;
      if (!CHECK || t2 < LOUT)
        hrow[(size_t)t2 * DD + co] = bf_bits(fmaxf(acc1[s][j] + bv, 0.f));
    }
  }
}

__global__ void __launch_bounds__(256) conv1_wmma_kernel(
    const unsigned short* __restrict__ xb,   // (32,2048,512) bf16
    const unsigned short* __restrict__ w1t,  // (512,20,512)  bf16 [co][k][ci]
    const float* __restrict__ bias,          // (512)
    unsigned short* __restrict__ hb)         // (32,2049,512) bf16 (relu'd)
{
  const int lane = threadIdx.x & 31;
  const int wave = threadIdx.x >> 5;
  const int wid  = blockIdx.x * 8 + wave;
  const int tilesPerBatch = 65 * 8;          // 65 t-tiles(32) x 8 co-tiles(64)
  const int b = wid / tilesPerBatch;
  if (b >= BB) return;
  const int rr  = wid - b * tilesPerBatch;
  const int t0  = (rr >> 3) * 32;
  const int co0 = (rr & 7) * 64;

  const int m    = lane & 15;                // A-row / B-col / D-col index
  const int half = lane >> 4;
  const int klo  = half << 3;

  const unsigned short* xbatch = xb + (size_t)b * LL * DD;
  unsigned short*       hrow   = hb + (size_t)b * LOUT * DD;

  // interior iff rows [t0-10, t0+40] all within [0,2048) and stores < 2049
  if (t0 >= 32 && t0 <= 1984)
    conv1_tile<false>(xbatch, w1t, bias, hrow, t0, co0, m, half, klo);
  else
    conv1_tile<true >(xbatch, w1t, bias, hrow, t0, co0, m, half, klo);
}

// ---------------- fused (rows x 512) @ (512 x 256) + relu, then tiny head ----
// mode 0: conv2 -> relu -> conv3 -> sigmoid   (out: 1 float / row)
// mode 1: lin1  -> relu -> lin2 -> softmax    (out: 4 floats / row)
__global__ void __launch_bounds__(128) gemm_n256_kernel(
    const unsigned short* __restrict__ A,      // (rows, 512) bf16
    const unsigned short* __restrict__ Wb,     // (256, 512) bf16, K contiguous
    const float* __restrict__ bias1,           // (256)
    const float* __restrict__ wsmall,          // mode0: (256); mode1: (4,256)
    const float* __restrict__ bsmall,          // mode0: (1);   mode1: (4)
    float* __restrict__ outp, int mode)
{
  __shared__ float sh[16][HD];
  __shared__ float red[16][8][RR];
  const int lane = threadIdx.x & 31;
  const int wave = threadIdx.x >> 5;           // 4 waves, each owns 64 columns
  const int m    = lane & 15;
  const int half = lane >> 4;
  const int klo  = half << 3;
  const long g0  = (long)blockIdx.x * 16;

  v8f acc[4];
  #pragma unroll
  for (int s = 0; s < 4; ++s) acc[s] = vz8();

  const unsigned short* arow = A + (size_t)(g0 + m) * DD + (half << 3);
  const unsigned short* wrow = Wb + (size_t)(wave * 64 + m) * DD + klo;
  #pragma unroll 2
  for (int ci0 = 0; ci0 < DD; ci0 += 32) {
    FragU a, f0, f1, f2, f3;
    a.u4[0] = *(const v4u*)(arow + ci0);
    a.u4[1] = *(const v4u*)(arow + ci0 + 16);
    const unsigned short* pW = wrow + ci0;
    ld_frag(f0, pW);
    ld_frag(f1, pW + 16 * DD);
    ld_frag(f2, pW + 32 * DD);
    ld_frag(f3, pW + 48 * DD);
    acc[0] = WMMA_BF16(a.bf, f0.bf, acc[0]);
    acc[1] = WMMA_BF16(a.bf, f1.bf, acc[1]);
    acc[2] = WMMA_BF16(a.bf, f2.bf, acc[2]);
    acc[3] = WMMA_BF16(a.bf, f3.bf, acc[3]);
  }

  #pragma unroll
  for (int s = 0; s < 4; ++s) {
    const int   co = wave * 64 + (s << 4) + m;
    const float bv = bias1[co];
    #pragma unroll
    for (int j = 0; j < 8; ++j)
      sh[j + (half << 3)][co] = fmaxf(acc[s][j] + bv, 0.f);
  }
  __syncthreads();

  // cooperative reduction: 128 threads = 16 positions x 8 slices of 32
  const int tid = threadIdx.x;
  const int p   = tid >> 3;
  const int c   = tid & 7;
  const int j0  = c * 32;
  if (mode == 0) {
    float s0 = 0.f;
    #pragma unroll 8
    for (int j = 0; j < 32; ++j) s0 += sh[p][j0 + j] * wsmall[j0 + j];
    red[p][c][0] = s0;
  } else {
    float s0 = 0.f, s1 = 0.f, s2 = 0.f, s3 = 0.f;
    #pragma unroll 4
    for (int j = 0; j < 32; ++j) {
      const float hv = sh[p][j0 + j];
      s0 += hv * wsmall[0 * HD + j0 + j];
      s1 += hv * wsmall[1 * HD + j0 + j];
      s2 += hv * wsmall[2 * HD + j0 + j];
      s3 += hv * wsmall[3 * HD + j0 + j];
    }
    red[p][c][0] = s0; red[p][c][1] = s1; red[p][c][2] = s2; red[p][c][3] = s3;
  }
  __syncthreads();

  if (tid < 16) {
    if (mode == 0) {
      float t = bsmall[0];
      #pragma unroll
      for (int c2 = 0; c2 < 8; ++c2) t += red[tid][c2][0];
      outp[g0 + tid] = 1.f / (1.f + expf(-t));
    } else {
      float l[RR];
      #pragma unroll
      for (int r = 0; r < RR; ++r) {
        float t = bsmall[r];
        #pragma unroll
        for (int c2 = 0; c2 < 8; ++c2) t += red[tid][c2][r];
        l[r] = t;
      }
      float mx = fmaxf(fmaxf(l[0], l[1]), fmaxf(l[2], l[3]));
      float e0 = expf(l[0] - mx), e1 = expf(l[1] - mx);
      float e2 = expf(l[2] - mx), e3 = expf(l[3] - mx);
      float inv = 1.f / (e0 + e1 + e2 + e3);
      float4 p4 = make_float4(e0 * inv, e1 * inv, e2 * inv, e3 * inv);
      *(float4*)(outp + (g0 + tid) * 4) = p4;
    }
  }
}

// ---------------- sequential regime scan (one thread per batch) -------------
#define SCH 16   // observations prefetched per chunk (2048 % 16 == 0)

__device__ __forceinline__ int argmax4(const float* v) {
  int bi = 0; float bv = v[0];
  #pragma unroll
  for (int i = 1; i < 4; ++i) if (v[i] > bv) { bv = v[i]; bi = i; }
  return bi;
}

__global__ void scan_kernel(const float* __restrict__ tp,
                            const float* __restrict__ rp,   // regime_probs
                            float* __restrict__ sm,         // smoothed
                            int*   __restrict__ asg,        // assignments
                            float* __restrict__ trans_out) {
  float tr[4][4];
  #pragma unroll
  for (int i = 0; i < 4; ++i) {
    float a = tp[i*4+0], b = tp[i*4+1], c = tp[i*4+2], d = tp[i*4+3];
    float mx = fmaxf(fmaxf(a, b), fmaxf(c, d));
    float e0 = expf(a-mx), e1 = expf(b-mx), e2 = expf(c-mx), e3 = expf(d-mx);
    float inv = 1.f / (e0 + e1 + e2 + e3);
    tr[i][0] = e0*inv; tr[i][1] = e1*inv; tr[i][2] = e2*inv; tr[i][3] = e3*inv;
  }
  const int b = threadIdx.x;
  if (b == 0)
    for (int i = 0; i < 16; ++i) trans_out[i] = tr[i >> 2][i & 3];
  if (b >= BB) return;

  const float* rpb = rp  + (size_t)b * LL * 4;
  float*       smb = sm  + (size_t)b * LL * 4;
  int*         ab  = asg + (size_t)b * LL;

  float prev[4] = {0.f, 0.f, 0.f, 0.f};

  for (int tb = 0; tb < LL; tb += SCH) {
    // batch-issue SCH independent loads, single wait
    float4 obs[SCH];
    #pragma unroll
    for (int i = 0; i < SCH; ++i)
      obs[i] = *(const float4*)(rpb + (size_t)(tb + i) * 4);
    // dependent recurrence on registers
    #pragma unroll
    for (int i = 0; i < SCH; ++i) {
      const int t = tb + i;
      float s[4];
      if (t == 0) {
        s[0] = obs[i].x; s[1] = obs[i].y; s[2] = obs[i].z; s[3] = obs[i].w;
      } else {
        const float ob[4] = {obs[i].x, obs[i].y, obs[i].z, obs[i].w};
        #pragma unroll
        for (int r = 0; r < 4; ++r) {
          float mm = prev[0]*tr[0][r] + prev[1]*tr[1][r]
                   + prev[2]*tr[2][r] + prev[3]*tr[3][r];
          s[r] = 0.7f * ob[r] + 0.3f * mm;
        }
      }
      *(float4*)(smb + (size_t)t * 4) = make_float4(s[0], s[1], s[2], s[3]);
      ab[t] = argmax4(s);
      #pragma unroll
      for (int r = 0; r < 4; ++r) prev[r] = s[r];
    }
  }
}

// ---------------- host launcher ----------------
extern "C" void kernel_launch(void* const* d_in, const int* in_sizes, int n_in,
                              void* d_out, int out_size, void* d_ws, size_t ws_size,
                              hipStream_t stream) {
  (void)in_sizes; (void)n_in; (void)out_size; (void)ws_size;

  const float* x    = (const float*)d_in[0];
  const float* w1   = (const float*)d_in[1];
  const float* b1   = (const float*)d_in[2];
  const float* w2   = (const float*)d_in[3];
  const float* b2   = (const float*)d_in[4];
  const float* w3   = (const float*)d_in[5];
  const float* b3   = (const float*)d_in[6];
  const float* l1w  = (const float*)d_in[7];
  const float* l1b  = (const float*)d_in[8];
  const float* l2w  = (const float*)d_in[9];
  const float* l2b  = (const float*)d_in[10];
  const float* tp   = (const float*)d_in[11];

  // workspace layout (bytes, 256-aligned)
  char* ws = (char*)d_ws;
  unsigned short* xb  = (unsigned short*)(ws);                         // 64 MB
  unsigned short* w1t = (unsigned short*)(ws + 67108864);              // 10 MB
  unsigned short* hb  = (unsigned short*)(ws + 77594624);              // 64.03 MB
  unsigned short* w2b = (unsigned short*)(ws + 144736256);             // 256 KB
  unsigned short* l1c = (unsigned short*)(ws + 144998400);             // 256 KB

  // output layout (floats, concatenated in return order)
  float* out      = (float*)d_out;
  float* o_change = out;                                   // 32*2049   = 65568
  float* o_rp     = out + 65568;                           // 32*2048*4 = 262144
  float* o_sm     = out + 65568 + 262144;
  int*   o_asg    = (int*)(out + 65568 + 262144 + 262144); // 32*2048
  float* o_trans  = out + 65568 + 262144 + 262144 + 65536; // 16

  // 1) precision conversion (f32 -> bf16)
  cvt_f32_bf16_kernel<<<32768, 256, 0, stream>>>(x,   xb,  (BB*LL*DD) / 4);
  cvt_w1_kernel      <<<20480, 256, 0, stream>>>(w1,  w1t);
  cvt_f32_bf16_kernel<<<128,   256, 0, stream>>>(w2,  w2b, (HD*DD) / 4);
  cvt_f32_bf16_kernel<<<128,   256, 0, stream>>>(l1w, l1c, (HD*DD) / 4);

  // 2) conv1 implicit GEMM (dominant 687 GFLOP) -> relu'd h (bf16)
  conv1_wmma_kernel<<<2080, 256, 0, stream>>>(xb, w1t, b1, hb);

  // 3) conv2(+relu)+conv3(+sigmoid): rows = 32*2049 = 65568 -> 4098 tiles
  gemm_n256_kernel<<<4098, 128, 0, stream>>>(hb, w2b, b2, w3, b3, o_change, 0);

  // 4) lin1(+relu)+lin2(+softmax): rows = 32*2048 = 65536 -> 4096 tiles
  gemm_n256_kernel<<<4096, 128, 0, stream>>>(xb, l1c, l1b, l2w, l2b, o_rp, 1);

  // 5) sequential smoothing scan + argmax + trans output
  scan_kernel<<<1, 32, 0, stream>>>(tp, o_rp, o_sm, o_asg, o_trans);
}
